// StandardGCNII_28759101014308
// MI455X (gfx1250) — matrix operand order repaired and checked by reading, not compiled, not run on hardware
//
#include <hip/hip_runtime.h>
#include <stdint.h>
#include <math.h>

// ---------------------------------------------------------------------------
// GCNII forward for MI455X (gfx1250, wave32, WMMA).
//
// Roofline: the 8x (adj_n @ h) GEMMs dominate: 68.7 GFLOP vs 8 passes over the
// adjacency. fp32 adjacency = 2.15 GB of HBM traffic and fp32 WMMA (16x16x4)
// can't cover 32 FLOP/byte; bf16 adjacency = 1.07 GB (~46us at 23.3 TB/s) and
// V_WMMA_F32_16X16X32_BF16 has ample throughput -> memory-bound, as desired.
// So: normalize+cast adj to bf16 once, keep h transposed in bf16 so WMMA A/B
// fragments are contiguous 16B lane loads, stage K-panels in LDS via
// global_load_async_to_lds_b128 (ASYNCcnt) with double buffering, accumulate
// in fp32, split-K=4 across workgroups into separate partial buffers.
//
// Round-2 refinements (from disasm feedback):
//  * preload all stage fragments into VGPRs, then chain the 8 WMMAs
//    back-to-back (kills the per-WMMA s_wait_dscnt 0x0 serialization)
//  * per-thread async src pointers precomputed once and bump-incremented,
//    removing the per-stage v_mad/v_lshl_add_u64 address recompute
// ---------------------------------------------------------------------------

typedef __bf16 bf16_t;
typedef __attribute__((ext_vector_type(8)))  __bf16 bf16x8;
typedef __attribute__((ext_vector_type(16))) __bf16 bf16x16;
typedef __attribute__((ext_vector_type(8)))  float  f32x8;

#define N_NODES 8192
#define FDIM    128
#define HDIM    64
#define CDIM    40
#define NLAYERS 8

#define TILE_M  128            // rows per workgroup (8 waves x 16 rows)
#define KSTEP   64             // K per LDS stage (2 WMMA k-iterations)
#define KPAD    (KSTEP + 8)    // +16B row pad -> conflict-free ds_load_b128
#define KSPLIT  4
#define KCHUNK  (N_NODES / KSPLIT)   // 2048
#define NSTAGES (KCHUNK / KSTEP)     // 32
// async b128 issued per thread per stage: A:4 + B:2 = 6
#define ASYNC_PER_STAGE 6

static __device__ __forceinline__ uint32_t lds32(const void* p) {
  // Low 32 bits of a flat pointer into LDS == LDS byte address.
  return (uint32_t)(uintptr_t)p;
}

static __device__ __forceinline__ void async_copy_b128(uint32_t lds, const void* g) {
  // GLOBAL_LOAD_ASYNC_TO_LDS_B128 (GV mode): vdst = LDS byte address VGPR,
  // vaddr = 64-bit global address. Tracked by ASYNCcnt.
  asm volatile("global_load_async_to_lds_b128 %0, %1, off"
               :: "v"(lds), "v"(g) : "memory");
}

static __device__ __forceinline__ bf16x16 cat8(bf16x8 lo, bf16x8 hi) {
  return __builtin_shufflevector(lo, hi, 0, 1, 2, 3, 4, 5, 6, 7, 8, 9, 10, 11,
                                 12, 13, 14, 15);
}

// --------------------------------------------------------------------------
// Kernel 1: d_inv_sqrt[i] = rsqrt(1 + sum_j adj[i][j])   (A+I row degree)
// --------------------------------------------------------------------------
__global__ __launch_bounds__(256) void rowsum_kernel(
    const float* __restrict__ adj, float* __restrict__ dinv) {
  __shared__ float red[256];
  const int row = blockIdx.x;
  const float* a = adj + (size_t)row * N_NODES;
  float s = 0.f;
  for (int j = threadIdx.x; j < N_NODES; j += 256) s += a[j];
  red[threadIdx.x] = s;
  __syncthreads();
  for (int w = 128; w > 0; w >>= 1) {
    if (threadIdx.x < w) red[threadIdx.x] += red[threadIdx.x + w];
    __syncthreads();
  }
  if (threadIdx.x == 0) dinv[row] = rsqrtf(red[0] + 1.0f);
}

// --------------------------------------------------------------------------
// Kernel 2: adjb[i][j] = bf16( d[i] * (adj[i][j] + (i==j)) * d[j] )
// --------------------------------------------------------------------------
__global__ __launch_bounds__(256) void norm_cast_kernel(
    const float* __restrict__ adj, const float* __restrict__ dinv,
    bf16_t* __restrict__ adjb) {
  size_t base = ((size_t)blockIdx.x * 256 + threadIdx.x) * 8;
  int i  = (int)(base >> 13);      // / 8192
  int j0 = (int)(base & 8191);
  float di = dinv[i];
  const float4* ap = (const float4*)(adj + base);
  float4 a0 = ap[0], a1 = ap[1];
  float v[8] = {a0.x, a0.y, a0.z, a0.w, a1.x, a1.y, a1.z, a1.w};
  bf16x8 o;
#pragma unroll
  for (int e = 0; e < 8; ++e) {
    int j = j0 + e;
    float val = v[e] + ((j == i) ? 1.0f : 0.0f);
    o[e] = (bf16_t)(di * dinv[j] * val);
  }
  *(bf16x8*)(adjb + base) = o;
}

// --------------------------------------------------------------------------
// Kernel 3: h0 = relu(x @ W_in + b_in); also emit hT0 (bf16, transposed)
// --------------------------------------------------------------------------
__global__ __launch_bounds__(256) void input_proj_kernel(
    const float* __restrict__ x, const float* __restrict__ Win,
    const float* __restrict__ bin, float* __restrict__ h0,
    bf16_t* __restrict__ hT0) {
  __shared__ float wL[FDIM * HDIM];   // 32 KB
  __shared__ float xL[16][FDIM];      // 8 KB
  const int tid = threadIdx.x;
  const int rowBase = blockIdx.x * 16;
  for (int i = tid; i < FDIM * HDIM; i += 256) wL[i] = Win[i];
  for (int i = tid; i < 16 * FDIM; i += 256)
    xL[i >> 7][i & 127] = x[(size_t)rowBase * FDIM + i];
  __syncthreads();
#pragma unroll
  for (int j = 0; j < 4; ++j) {
    int o = tid + 256 * j;                // 16 rows x 64 cols = 1024 outputs
    int r = o >> 6, n = o & 63;
    float acc = bin[n];
#pragma unroll 8
    for (int k = 0; k < FDIM; ++k) acc += xL[r][k] * wL[k * HDIM + n];
    acc = fmaxf(acc, 0.f);
    int row = rowBase + r;
    h0[(size_t)row * HDIM + n] = acc;
    hT0[(size_t)n * N_NODES + row] = (bf16_t)acc;
  }
}

// --------------------------------------------------------------------------
// Kernel 4: h_part[ksplit] = adj_n[:, kchunk] @ h[kchunk, :]   (bf16 WMMA)
//   grid = (N/TILE_M, KSPLIT), block = 256 (8 waves; 16 rows x 64 cols each)
// --------------------------------------------------------------------------
__global__ __launch_bounds__(256) void gcn_prop_kernel(
    const bf16_t* __restrict__ adjb, const bf16_t* __restrict__ hTin,
    float* __restrict__ hpart) {
  __shared__ alignas(16) bf16_t As[2][TILE_M][KPAD];  // 2*128*72*2 = 36 KB
  __shared__ alignas(16) bf16_t Bs[2][HDIM][KPAD];    // 2* 64*72*2 = 18 KB
  const int tid = threadIdx.x;
  const int mBase = blockIdx.x * TILE_M;
  const int kOrigin = blockIdx.y * KCHUNK;
  float* __restrict__ outp = hpart + (size_t)blockIdx.y * N_NODES * HDIM;

  const uint32_t asB[2] = {lds32(&As[0][0][0]), lds32(&As[1][0][0])};
  const uint32_t bsB[2] = {lds32(&Bs[0][0][0]), lds32(&Bs[1][0][0])};

  // Per-thread async source pointers (bump-incremented by KSTEP per stage)
  // and fixed LDS byte offsets within a buffer.
  const bf16_t* aptr[4];
  uint32_t aoff[4];
#pragma unroll
  for (int j = 0; j < 4; ++j) {   // A: 128 rows x 8 chunks of 16B
    int u = tid * 4 + j;
    int row = u >> 3, ch = u & 7;
    aptr[j] = adjb + (size_t)(mBase + row) * N_NODES + kOrigin + ch * 8;
    aoff[j] = (uint32_t)(row * KPAD + ch * 8) * 2;
  }
  const bf16_t* bptr[2];
  uint32_t boff[2];
#pragma unroll
  for (int j = 0; j < 2; ++j) {   // B: 64 hT rows x 8 chunks of 16B
    int u = tid * 2 + j;
    int n = u >> 3, ch = u & 7;
    bptr[j] = hTin + (size_t)n * N_NODES + kOrigin + ch * 8;
    boff[j] = (uint32_t)(n * KPAD + ch * 8) * 2;
  }

  // Prologue: stage 0 -> buffer 0.
#pragma unroll
  for (int j = 0; j < 4; ++j) {
    async_copy_b128(asB[0] + aoff[j], aptr[j]);
    aptr[j] += KSTEP;
  }
#pragma unroll
  for (int j = 0; j < 2; ++j) {
    async_copy_b128(bsB[0] + boff[j], bptr[j]);
    bptr[j] += KSTEP;
  }

  const int lane = tid & 31;
  const int wave = tid >> 5;
  const int r  = lane & 15;     // A: row-in-slab / B: column-in-tile
  const int hi = lane >> 4;     // ISA half-wave K interleave select
  const int rowOff = wave * 16;

  f32x8 acc[4] = {};            // 4 n-tiles of 16x16 fp32

#pragma unroll 1
  for (int s = 0; s < NSTAGES; ++s) {
    if (s + 1 < NSTAGES) {
      const int nb = (s + 1) & 1;
#pragma unroll
      for (int j = 0; j < 4; ++j) {
        async_copy_b128(asB[nb] + aoff[j], aptr[j]);
        aptr[j] += KSTEP;
      }
#pragma unroll
      for (int j = 0; j < 2; ++j) {
        async_copy_b128(bsB[nb] + boff[j], bptr[j]);
        bptr[j] += KSTEP;
      }
      asm volatile("s_wait_asynccnt 0x6" ::: "memory");  // <= ASYNC_PER_STAGE
    } else {
      asm volatile("s_wait_asynccnt 0x0" ::: "memory");
    }
    __syncthreads();
    const int buf = s & 1;

    // Preload ALL fragments for this stage (2 A + 8 B), then chain WMMAs.
    bf16x16 afrag[2];
    bf16x16 bfrag[2][4];
#pragma unroll
    for (int k2 = 0; k2 < 2; ++k2) {
      const int kk = k2 * 32;
      // A fragment 16x32 bf16: lanes 0-15 -> K kk+0..7 / kk+16..23,
      //                        lanes 16-31 -> K kk+8..15 / kk+24..31
      bf16x8 alo = *(const bf16x8*)&As[buf][rowOff + r][kk + hi * 8];
      bf16x8 ahi = *(const bf16x8*)&As[buf][rowOff + r][kk + 16 + hi * 8];
      afrag[k2] = cat8(alo, ahi);
#pragma unroll
      for (int nt = 0; nt < 4; ++nt) {
        // B fragment 32x16: lane column = r, lanes 0-15 K kk+0..7 / kk+16..23,
        //                   lanes 16-31 K kk+8..15 / kk+24..31
        bf16x8 blo = *(const bf16x8*)&Bs[buf][nt * 16 + r][kk + hi * 8];
        bf16x8 bhi = *(const bf16x8*)&Bs[buf][nt * 16 + r][kk + 16 + hi * 8];
        bfrag[k2][nt] = cat8(blo, bhi);
      }
    }
#pragma unroll
    for (int k2 = 0; k2 < 2; ++k2)
#pragma unroll
      for (int nt = 0; nt < 4; ++nt)
        acc[nt] = __builtin_amdgcn_wmma_f32_16x16x32_bf16(
            false, afrag[k2], false, bfrag[k2][nt], (short)0, acc[nt], false,
            false);
    __syncthreads();
  }

  // D layout: VGPR g, lane L -> (M = g + 8*(L>>4), N = L&15)
#pragma unroll
  for (int nt = 0; nt < 4; ++nt)
#pragma unroll
    for (int g = 0; g < 8; ++g) {
      int m = mBase + rowOff + hi * 8 + g;
      int n = nt * 16 + r;
      outp[(size_t)m * HDIM + n] = acc[nt][g];
    }
}

// --------------------------------------------------------------------------
// Kernel 5: per-layer epilogue:
//   h_mixed = 0.9*(sum_p h_part[p]) + 0.1*h0
//   hT_out  = bf16( relu(h_mixed @ ((1-beta)I + beta*W) + b) )^T
// --------------------------------------------------------------------------
__global__ __launch_bounds__(256) void gcn_epilogue_kernel(
    const float* __restrict__ hpart, const float* __restrict__ h0,
    const float* __restrict__ cw, const float* __restrict__ cb, float beta,
    bf16_t* __restrict__ hTo) {
  __shared__ float wL[HDIM][HDIM + 1];
  __shared__ float hm[HDIM][HDIM + 1];
  const int tid = threadIdx.x;
  const int rowBase = blockIdx.x * 64;
  const size_t P = (size_t)N_NODES * HDIM;
#pragma unroll
  for (int i = 0; i < 16; ++i) {
    int idx = tid + i * 256;
    int k = idx >> 6, n = idx & 63;
    float w = beta * cw[idx];
    if (k == n) w += (1.0f - beta);
    wL[k][n] = w;
    size_t e = (size_t)(rowBase + k) * HDIM + n;
    float hp = hpart[e] + hpart[e + P] + hpart[e + 2 * P] + hpart[e + 3 * P];
    hm[k][n] = 0.9f * hp + 0.1f * h0[e];
  }
  __syncthreads();
#pragma unroll
  for (int i = 0; i < 16; ++i) {
    int idx = tid + i * 256;
    int r = idx >> 6, n = idx & 63;
    float acc = cb[n];
#pragma unroll 8
    for (int k = 0; k < HDIM; ++k) acc += hm[r][k] * wL[k][n];
    acc = fmaxf(acc, 0.f);
    hTo[(size_t)n * N_NODES + rowBase + r] = (bf16_t)acc;
  }
}

// --------------------------------------------------------------------------
// Kernel 6: logits = h @ W_out + b_out; out = log_softmax(logits)
// --------------------------------------------------------------------------
__global__ __launch_bounds__(256) void out_proj_kernel(
    const bf16_t* __restrict__ hT, const float* __restrict__ Wout,
    const float* __restrict__ bout, float* __restrict__ out) {
  __shared__ float wL[HDIM * CDIM];
  __shared__ float bL[CDIM];
  __shared__ bf16_t hL[HDIM][256];
  const int tid = threadIdx.x;
  const int rowBase = blockIdx.x * 256;
  for (int i = tid; i < HDIM * CDIM; i += 256) wL[i] = Wout[i];
  if (tid < CDIM) bL[tid] = bout[tid];
#pragma unroll
  for (int k = 0; k < HDIM; ++k)
    hL[k][tid] = hT[(size_t)k * N_NODES + rowBase + tid];
  __syncthreads();
  float lg[CDIM];
#pragma unroll
  for (int n = 0; n < CDIM; ++n) lg[n] = bL[n];
  for (int k = 0; k < HDIM; ++k) {
    float hv = (float)hL[k][tid];
#pragma unroll
    for (int n = 0; n < CDIM; ++n) lg[n] += hv * wL[k * CDIM + n];
  }
  float mx = lg[0];
#pragma unroll
  for (int n = 1; n < CDIM; ++n) mx = fmaxf(mx, lg[n]);
  float s = 0.f;
#pragma unroll
  for (int n = 0; n < CDIM; ++n) s += __expf(lg[n] - mx);
  float lse = mx + __logf(s);
  size_t row = (size_t)rowBase + tid;
#pragma unroll
  for (int n = 0; n < CDIM; ++n) out[row * CDIM + n] = lg[n] - lse;
}

// --------------------------------------------------------------------------
extern "C" void kernel_launch(void* const* d_in, const int* in_sizes, int n_in,
                              void* d_out, int out_size, void* d_ws,
                              size_t ws_size, hipStream_t stream) {
  (void)in_sizes; (void)n_in; (void)out_size; (void)ws_size;
  const float* x     = (const float*)d_in[0];
  const float* adj   = (const float*)d_in[1];
  const float* W_in  = (const float*)d_in[2];
  const float* b_in  = (const float*)d_in[3];
  const float* convw = (const float*)d_in[4];
  const float* convb = (const float*)d_in[5];
  const float* W_out = (const float*)d_in[6];
  const float* b_out = (const float*)d_in[7];
  float* out = (float*)d_out;

  // Workspace layout (~147 MB): all chunks 256B-aligned.
  char* ws = (char*)d_ws;
  bf16_t* adjb = (bf16_t*)ws;  ws += (size_t)N_NODES * N_NODES * sizeof(bf16_t);
  float*  dinv = (float*)ws;   ws += (size_t)N_NODES * sizeof(float);
  float*  h0   = (float*)ws;   ws += (size_t)N_NODES * HDIM * sizeof(float);
  bf16_t* hT0  = (bf16_t*)ws;  ws += (size_t)HDIM * N_NODES * sizeof(bf16_t);
  bf16_t* hT1  = (bf16_t*)ws;  ws += (size_t)HDIM * N_NODES * sizeof(bf16_t);
  float*  hpart = (float*)ws;  // KSPLIT * N * H floats

  rowsum_kernel<<<N_NODES, 256, 0, stream>>>(adj, dinv);
  norm_cast_kernel<<<(N_NODES * (size_t)N_NODES / 8) / 256, 256, 0, stream>>>(
      adj, dinv, adjb);
  input_proj_kernel<<<N_NODES / 16, 256, 0, stream>>>(x, W_in, b_in, h0, hT0);

  bf16_t* hin = hT0;
  bf16_t* hout = hT1;
  for (int i = 0; i < NLAYERS; ++i) {
    gcn_prop_kernel<<<dim3(N_NODES / TILE_M, KSPLIT), 256, 0, stream>>>(
        adjb, hin, hpart);
    float beta = logf(0.5f / (float)(i + 1) + 1.0f);
    gcn_epilogue_kernel<<<N_NODES / 64, 256, 0, stream>>>(
        hpart, h0, convw + (size_t)i * HDIM * HDIM, convb + (size_t)i * HDIM,
        beta, hout);
    bf16_t* t = hin; hin = hout; hout = t;
  }
  out_proj_kernel<<<N_NODES / 256, 256, 0, stream>>>(hin, W_out, b_out, out);
}
